// SelfAttention_32607391711321
// MI455X (gfx1250) — matrix operand order repaired and checked by reading, not compiled, not run on hardware
//
#include <hip/hip_runtime.h>
#include <stdint.h>

typedef __attribute__((ext_vector_type(16))) _Float16 v16h;
typedef __attribute__((ext_vector_type(8)))  float    v8f;

#define B_  4
#define C_  256
#define CQK 32
#define N_  4096

// Workspace partition (units: _Float16 elements).
//  qA: per (b, i-tile) 16x32 f16 A-operand tiles : 4*256*32lanes*16halfs = 524288
//  kB: per (b, j-tile) 32x16 f16 B-operand tiles : 524288
//  vB: per (b, j-chunk, c-tile) 32x16 f16 B-operand tiles : 4*128*16*32*16 = 4194304
// total 5 Mi halfs = 10 MiB
#define QA_OFF  ((size_t)0)
#define KB_OFF  ((size_t)B_ * 256 * 32 * 16)
#define VB_OFF  (KB_OFF * 2)

// ---------------------------------------------------------------------------
// Projection: q = Wq x + bq, k = Wk x + bk, v = Wv x + bv, written as f16 in
// WMMA-operand-swizzled layouts. Each thread: one spatial n, 16 out channels.
// Output chunks: oc 0-1 -> q(32), oc 2-3 -> k(32), oc 4-19 -> v(256).
// ---------------------------------------------------------------------------
__global__ void __launch_bounds__(256) proj_swizzle_kernel(
    const float* __restrict__ x,
    const float* __restrict__ Wq, const float* __restrict__ bq,
    const float* __restrict__ Wk, const float* __restrict__ bk,
    const float* __restrict__ Wv, const float* __restrict__ bv,
    _Float16* __restrict__ ws)
{
    const int bid = blockIdx.x;
    const int n   = ((bid & 15) << 8) + threadIdx.x;   // 0..4095
    const int oc  = (bid >> 4) % 20;                   // 16-channel chunk
    const int b   = bid / 320;

    const float* __restrict__ Wrow;
    const float* __restrict__ bias;
    if (oc < 2)      { Wrow = Wq + (size_t)oc * 16 * C_;       bias = bq + oc * 16; }
    else if (oc < 4) { Wrow = Wk + (size_t)(oc - 2) * 16 * C_; bias = bk + (oc - 2) * 16; }
    else             { Wrow = Wv + (size_t)(oc - 4) * 16 * C_; bias = bv + (oc - 4) * 16; }

    const float* __restrict__ xb = x + (size_t)b * C_ * N_ + n;

    float acc[16];
#pragma unroll
    for (int i = 0; i < 16; ++i) acc[i] = bias[i];

    for (int c = 0; c < C_; ++c) {
        const float xv = xb[(size_t)c * N_];
#pragma unroll
        for (int i = 0; i < 16; ++i)
            acc[i] = __builtin_fmaf(Wrow[i * C_ + c], xv, acc[i]);
    }

#pragma unroll
    for (int i = 0; i < 16; ++i) {
        const _Float16 hv = (_Float16)acc[i];
        size_t idx;
        if (oc < 2) {                      // q -> A-operand (M=i rows, K=channel)
            const int K  = oc * 16 + i;
            const int g  = (K >> 4) & 1;
            const int h  = K & 15;
            const int it = n >> 4;
            const int L  = (n & 15) + 16 * g;
            idx = QA_OFF + ((((size_t)b * 256 + it) * 32 + L) * 16 + h);
        } else if (oc < 4) {               // k -> B-operand (K=channel, N=j)
            const int ck = (oc - 2) * 16 + i;
            const int jt = n >> 4;
            const int L  = (n & 15) + 16 * (ck >= 16);
            const int h  = ck & 15;
            idx = KB_OFF + ((((size_t)b * 256 + jt) * 32 + L) * 16 + h);
        } else {                           // v -> B-operand (K=j, N=c)
            const int ct = oc - 4;         // c / 16 ; c % 16 == i
            const int jc = n >> 5;
            const int L  = i + 16 * ((n >> 4) & 1);
            const int h  = n & 15;
            idx = VB_OFF + (((((size_t)b * 128 + jc) * 16 + ct) * 32 + L) * 16 + h);
        }
        ws[idx] = hv;
    }
}

// ---------------------------------------------------------------------------
// Flash attention: 1 wave = one 16-row i-tile; 4 waves / workgroup.
// Pass 1: fused online softmax statistics (max + sum, shuffle-free in-loop).
// Pass 2: O = P * V^T with async-double-buffered V tiles in LDS.
// ---------------------------------------------------------------------------
__global__ void __launch_bounds__(128) flash_attn_kernel(
    const _Float16* __restrict__ ws,
    const float* __restrict__ x,
    const float* __restrict__ gamma,
    float* __restrict__ out)
{
    const _Float16* __restrict__ qA = ws + QA_OFF;
    const _Float16* __restrict__ kB = ws + KB_OFF;
    const _Float16* __restrict__ vB = ws + VB_OFF;

    const int wg   = blockIdx.x;        // 0..255
    const int b    = wg >> 6;
    const int rb   = wg & 63;
    const int wave = threadIdx.x >> 5;
    const int lane = threadIdx.x & 31;
    const int it   = rb * 4 + wave;     // i-tile (16 query rows)
    const int g    = lane >> 4;
    const int ln   = lane & 15;

    __shared__ alignas(32) _Float16 vbuf[2][16 * 32 * 16]; // 2 x 16 KiB: V B-tiles
    __shared__ alignas(32) _Float16 pbuf[4 * 16 * 32];     // 4 KiB: per-wave scratch

    // Loop-invariant A operand (q tile), contiguous 32 B per lane.
    const v16h qa = *(const v16h*)(qA + (((size_t)b * 256 + it) * 32 + lane) * 16);
    const _Float16* __restrict__ kbb = kB + (size_t)b * 256 * 512 + (size_t)lane * 16;

    // -------- Pass 1: fused online softmax stats over all 4096 keys --------
    float m[8], s[8];
#pragma unroll
    for (int r = 0; r < 8; ++r) { m[r] = -1e30f; s[r] = 0.f; }

    for (int jt = 0; jt < 256; ++jt) {
        if (jt + 8 < 256) __builtin_prefetch(kbb + (size_t)(jt + 8) * 512, 0, 1);
        const v16h kb = *(const v16h*)(kbb + (size_t)jt * 512);
        v8f z = {0.f, 0.f, 0.f, 0.f, 0.f, 0.f, 0.f, 0.f};
        v8f sc = __builtin_amdgcn_wmma_f32_16x16x32_f16(
            false, qa, false, kb, (short)0, z, false, false);
#pragma unroll
        for (int r = 0; r < 8; ++r) {
            const float mn = fmaxf(m[r], sc[r]);
            s[r] = s[r] * __expf(m[r] - mn) + __expf(sc[r] - mn);
            m[r] = mn;
        }
    }
    // combine (m,s) pairs across the 16-lane group (a C-row spans 16 lanes)
    float inv[8];
#pragma unroll
    for (int r = 0; r < 8; ++r) {
#pragma unroll
        for (int off = 1; off < 16; off <<= 1) {
            const float mo = __shfl_xor(m[r], off, 16);
            const float so = __shfl_xor(s[r], off, 16);
            const float mn = fmaxf(m[r], mo);
            s[r] = s[r] * __expf(m[r] - mn) + so * __expf(mo - mn);
            m[r] = mn;
        }
        inv[r] = 1.0f / s[r];
    }

    // -------- Pass 2: O = P * V^T, async double-buffered V staging --------
    v8f acc[16];
    {
        v8f z = {0.f, 0.f, 0.f, 0.f, 0.f, 0.f, 0.f, 0.f};
#pragma unroll
        for (int ct = 0; ct < 16; ++ct) acc[ct] = z;
    }

    const char* __restrict__ vbase =
        (const char*)(vB + (size_t)b * 128 * 16 * 32 * 16);   // + jc*16384 bytes

    // issue one 16 KiB chunk as 8 async 16B transfers per thread
    auto issue_chunk = [&](int jc, int buf) {
        const char* g0 = vbase + (size_t)jc * 16384;
        const unsigned l0 = (unsigned)(uintptr_t)&vbuf[buf][0];
#pragma unroll
        for (int k = 0; k < 8; ++k) {
            const unsigned boff = ((unsigned)threadIdx.x + 128u * k) * 16u;
            asm volatile("global_load_async_to_lds_b128 %0, %1, off"
                         :: "v"(l0 + boff), "v"(g0 + boff) : "memory");
        }
    };

    issue_chunk(0, 0);
    asm volatile("s_wait_asynccnt 0x0" ::: "memory");
    __syncthreads();

    for (int jc = 0; jc < 128; ++jc) {
        const int cur = jc & 1;
        if (jc + 1 < 128) issue_chunk(jc + 1, cur ^ 1);   // overlap with WMMAs below

        // Scores for the two 16-key sub-tiles; normalized P -> LDS (M,K order).
#pragma unroll
        for (int sub = 0; sub < 2; ++sub) {
            const int jt = jc * 2 + sub;
            const v16h kb = *(const v16h*)(kbb + (size_t)jt * 512);
            v8f z = {0.f, 0.f, 0.f, 0.f, 0.f, 0.f, 0.f, 0.f};
            v8f sc = __builtin_amdgcn_wmma_f32_16x16x32_f16(
                false, qa, false, kb, (short)0, z, false, false);
#pragma unroll
            for (int r = 0; r < 8; ++r) {
                const float p = __expf(sc[r] - m[r]) * inv[r];
                pbuf[wave * 512 + (r + 8 * g) * 32 + ln + 16 * sub] = (_Float16)p;
            }
        }

        // pbuf is wave-private: wave-local DS completion is sufficient.
        asm volatile("s_wait_dscnt 0x0" ::: "memory");

        // Repack P into the f16 A-operand layout (K-consecutive half pairs).
        union { v16h h; unsigned u[8]; } pa;
#pragma unroll
        for (int p2 = 0; p2 < 8; ++p2) {
            const int K0 = 2 * p2 + 16 * g;
            pa.u[p2] = *(const unsigned*)&pbuf[wave * 512 + ln * 32 + K0];
        }

        // 16 accumulate WMMAs covering all 256 value channels.
#pragma unroll
        for (int ct = 0; ct < 16; ++ct) {
            const v16h vb = *(const v16h*)&vbuf[cur][(ct * 32 + lane) * 16];
            acc[ct] = __builtin_amdgcn_wmma_f32_16x16x32_f16(
                false, pa.h, false, vb, (short)0, acc[ct], false, false);
        }

        // Next buffer must be fully in LDS and visible to all waves; also
        // guarantees everyone is done reading vbuf[cur] before it is reused.
        asm volatile("s_wait_asynccnt 0x0" ::: "memory");
        __syncthreads();
    }

    // -------- Epilogue: out = gamma*O + x, coalesced via LDS transpose --------
    const float gam = gamma[0];
    const int i0 = it * 16;
    float* tb = (float*)&pbuf[(size_t)wave * 512];   // 16x16 f32 tile [c][i]
#pragma unroll
    for (int ct = 0; ct < 16; ++ct) {
#pragma unroll
        for (int r = 0; r < 8; ++r)
            tb[ln * 16 + (r + 8 * g)] = acc[ct][r];   // T[c=ln][i=r+8g]
        asm volatile("s_wait_dscnt 0x0" ::: "memory");
#pragma unroll
        for (int p = 0; p < 8; ++p) {
            const int cw = 2 * p + g;                 // two 64B rows per store
            const size_t o = ((size_t)b * C_ + ct * 16 + cw) * (size_t)N_ + i0 + ln;
            out[o] = __builtin_fmaf(gam, tb[cw * 16 + ln], x[o]);
        }
        asm volatile("" ::: "memory");                // reads before next tile's writes
    }
}

// ---------------------------------------------------------------------------
extern "C" void kernel_launch(void* const* d_in, const int* in_sizes, int n_in,
                              void* d_out, int out_size, void* d_ws, size_t ws_size,
                              hipStream_t stream)
{
    (void)in_sizes; (void)n_in; (void)out_size; (void)ws_size;
    const float* x     = (const float*)d_in[0];
    const float* Wq    = (const float*)d_in[1];
    const float* bq    = (const float*)d_in[2];
    const float* Wk    = (const float*)d_in[3];
    const float* bk    = (const float*)d_in[4];
    const float* Wv    = (const float*)d_in[5];
    const float* bv    = (const float*)d_in[6];
    const float* gamma = (const float*)d_in[7];
    _Float16* ws = (_Float16*)d_ws;
    float* out = (float*)d_out;

    // 4 batches * 20 channel-chunks * 16 n-chunks
    proj_swizzle_kernel<<<dim3(B_ * 20 * 16), dim3(256), 0, stream>>>(
        x, Wq, bq, Wk, bk, Wv, bv, ws);
    // 4 batches * 64 row-blocks (64 query rows each, 4 waves of 16)
    flash_attn_kernel<<<dim3(B_ * 64), dim3(128), 0, stream>>>(ws, x, gamma, out);
}
